// LabelAttention_51548197487425
// MI455X (gfx1250) — matrix operand order repaired and checked by reading, not compiled, not run on hardware
//
#include <hip/hip_runtime.h>
#include <hip/hip_bf16.h>
#include <math.h>

// Problem constants (match reference)
#define BDIM 8
#define SDIM 512
#define HDIM 768
#define LDIM 8192
#define MT   32   // labels per block in the fused attention kernel

typedef _Float16 f16x4 __attribute__((ext_vector_type(4)));
typedef _Float16 f16x8 __attribute__((ext_vector_type(8)));
typedef _Float16 v16h  __attribute__((ext_vector_type(16)));
typedef float    v8f   __attribute__((ext_vector_type(8)));

union AV { v16h v; f16x8 h[2]; };
union CV { v8f  v; float f[8]; };

// ---------------------------------------------------------------------------
// WMMA operand loaders (wave32, V_WMMA_F32_16X16X32_F16 layouts)
//   A (16xK tile, row-major [row][K], ld in halves):
//     lane l: row = l&15, hf = l>>4
//     elems 0..7  = K[k0 + hf*8 .. +7],  elems 8..15 = K[k0+16+hf*8 .. +7]
//   B ([N][K] row-major == B^T of the KxN matrix):
//     lane l: col = l&15, hf = l>>4
//     elems 0..7  = K[k0 + hf*8 .. +7],  elems 8..15 = K[k0+16+hf*8 .. +7]
// ---------------------------------------------------------------------------
__device__ __forceinline__ v16h load_matA(const _Float16* base, int ld, int k0, int lane) {
    int r  = lane & 15;
    int hf = lane >> 4;
    const _Float16* p = base + (size_t)r * ld + k0 + hf * 8;
    AV a;
    a.h[0] = *(const f16x8*)(p);
    a.h[1] = *(const f16x8*)(p + 16);
    return a.v;
}

__device__ __forceinline__ v16h load_matB(const _Float16* base, int ld, int k0, int lane) {
    int c  = lane & 15;
    int hf = lane >> 4;
    const _Float16* p = base + (size_t)c * ld + k0 + hf * 8;
    AV b;
    b.h[0] = *(const f16x8*)(p);
    b.h[1] = *(const f16x8*)(p + 16);
    return b.v;
}

__device__ __forceinline__ v8f wmma_f16(v16h a, v16h b, v8f c) {
    return __builtin_amdgcn_wmma_f32_16x16x32_f16(false, a, false, b, (short)0, c,
                                                  false, false);
}

// ---------------------------------------------------------------------------
// fp32 -> f16 elementwise cast (n4 = n/4)
// ---------------------------------------------------------------------------
__global__ void __launch_bounds__(256) cvt_f32_f16(const float* __restrict__ in,
                                                   _Float16* __restrict__ out, int n4) {
    int i = (blockIdx.x * 256 + threadIdx.x);
    if (i >= n4) return;
    float4 v = *(const float4*)(in + (size_t)i * 4);
    f16x4 o = { (_Float16)v.x, (_Float16)v.y, (_Float16)v.z, (_Float16)v.w };
    *(f16x4*)(out + (size_t)i * 4) = o;
}

// ---------------------------------------------------------------------------
// inputs [B,S,H] fp32 -> X16T [B,H,S] f16 (transposed copy for GEMM3 B-operand)
// ---------------------------------------------------------------------------
__global__ void __launch_bounds__(256) transpose_f16(const float* __restrict__ in,
                                                     _Float16* __restrict__ out) {
    size_t idx = (size_t)blockIdx.x * 256 + threadIdx.x;  // over B*H*S
    int s = (int)(idx % SDIM);
    size_t r = idx / SDIM;
    int h = (int)(r % HDIM);
    int b = (int)(r / HDIM);
    out[idx] = (_Float16)in[((size_t)b * SDIM + s) * HDIM + h];
}

// ---------------------------------------------------------------------------
// GEMM1: K16[b,s,k] = sum_h X16[b,s,h] * W16[k,h]     (attn_key = X @ W^T)
// grid = B * S/16 blocks of 256 threads (8 waves); wave w owns N cols [w*96, +96)
// ---------------------------------------------------------------------------
__global__ void __launch_bounds__(256) gemm_key(const _Float16* __restrict__ X16,
                                                const _Float16* __restrict__ W16,
                                                _Float16* __restrict__ K16) {
    __shared__ __align__(16) _Float16 Atile[16 * HDIM];   // 24 KB
    int b  = blockIdx.x >> 5;          // / (S/16 == 32)
    int s0 = (blockIdx.x & 31) << 4;
    int tid = threadIdx.x;

    const _Float16* src = X16 + ((size_t)b * SDIM + s0) * HDIM;
    for (int i = tid; i < (16 * HDIM) / 8; i += 256)
        *(f16x8*)&Atile[i * 8] = *(const f16x8*)&src[i * 8];
    __syncthreads();

    int wave = tid >> 5, lane = tid & 31;
    int col = lane & 15, hf = lane >> 4;
    int n0 = wave * 96;

    CV c[6];
    #pragma unroll
    for (int t = 0; t < 6; ++t)
        #pragma unroll
        for (int v = 0; v < 8; ++v) c[t].f[v] = 0.0f;

    for (int k0 = 0; k0 < HDIM; k0 += 32) {
        v16h a = load_matA(Atile, HDIM, k0, lane);
        #pragma unroll
        for (int t = 0; t < 6; ++t) {
            v16h bm = load_matB(W16 + (size_t)(n0 + t * 16) * HDIM, HDIM, k0, lane);
            c[t].v = wmma_f16(a, bm, c[t].v);
        }
    }

    #pragma unroll
    for (int t = 0; t < 6; ++t)
        #pragma unroll
        for (int v = 0; v < 8; ++v) {
            int m = v + hf * 8;
            int n = n0 + t * 16 + col;
            K16[((size_t)b * SDIM + s0 + m) * HDIM + n] = (_Float16)c[t].f[v];
        }
}

// ---------------------------------------------------------------------------
// Fused: scores = E_tile @ K16[b]^T -> mask -> softmax(S) -> out = P @ X
// grid = B * L/32 blocks of 256 threads (8 waves). 32 labels per block so each
// B-operand fetch (L2) feeds TWO wmmas -> halved L2 traffic per FLOP.
// Phase1: wave w owns score cols [w*64,+64). Phase2: wave w owns H cols [w*96,+96).
// ---------------------------------------------------------------------------
__global__ void __launch_bounds__(256) label_attn(const _Float16* __restrict__ L16,
                                                  const _Float16* __restrict__ K16,
                                                  const _Float16* __restrict__ X16T,
                                                  const unsigned char* __restrict__ masks,
                                                  float* __restrict__ out) {
    // 48 KB region: label tile (32x768 f16), later reused as scores/P (32x512 f16)
    __shared__ __align__(16) unsigned char smemA[MT * HDIM * 2];
    __shared__ float red[MT][8];
    __shared__ float rowmax[MT];
    __shared__ float rowsum[MT];
    const float NEG_INF = -__builtin_inff();

    int b  = blockIdx.x >> 8;          // / (L/32 == 256)
    int l0 = (blockIdx.x & 255) << 5;
    int tid = threadIdx.x;
    _Float16* Alab = (_Float16*)smemA;

    // stage 32x768 f16 label tile (contiguous rows) into LDS
    const _Float16* src = L16 + (size_t)l0 * HDIM;
    for (int i = tid; i < (MT * HDIM) / 8; i += 256)
        *(f16x8*)&Alab[i * 8] = *(const f16x8*)&src[i * 8];
    __syncthreads();

    int wave = tid >> 5, lane = tid & 31;
    int col = lane & 15, hf = lane >> 4;

    // ---- Phase 1: score strip 32 x 64 per wave, K = H = 768 ----
    int n0 = wave * 64;
    CV c0[4], c1[4];
    #pragma unroll
    for (int t = 0; t < 4; ++t)
        #pragma unroll
        for (int v = 0; v < 8; ++v) { c0[t].f[v] = 0.0f; c1[t].f[v] = 0.0f; }

    const _Float16* Kb = K16 + (size_t)b * SDIM * HDIM;
    for (int k0 = 0; k0 < HDIM; k0 += 32) {
        v16h a0 = load_matA(Alab, HDIM, k0, lane);
        v16h a1 = load_matA(Alab + (size_t)16 * HDIM, HDIM, k0, lane);
        #pragma unroll
        for (int t = 0; t < 4; ++t) {
            v16h bm = load_matB(Kb + (size_t)(n0 + t * 16) * HDIM, HDIM, k0, lane);
            c0[t].v = wmma_f16(a0, bm, c0[t].v);
            c1[t].v = wmma_f16(a1, bm, c1[t].v);
        }
    }
    __syncthreads();   // all waves done reading Alab; region is reusable

    // mask + spill f16 scores (in the former label-tile region)
    _Float16* Psc = (_Float16*)smemA;          // [32][512] f16, 32 KB
    const _Float16 HNEG = (_Float16)NEG_INF;   // f16 -inf
    const unsigned char* mb = masks + (size_t)b * SDIM;
    #pragma unroll
    for (int t = 0; t < 4; ++t) {
        int n = n0 + t * 16 + col;
        bool ok = (mb[n] != 0);
        #pragma unroll
        for (int v = 0; v < 8; ++v) {
            int m = v + hf * 8;
            Psc[(size_t)m * SDIM + n]        = ok ? (_Float16)c0[t].f[v] : HNEG;
            Psc[(size_t)(m + 16) * SDIM + n] = ok ? (_Float16)c1[t].f[v] : HNEG;
        }
    }
    __syncthreads();

    // ---- masked softmax over S (32 rows x 512): 8 threads per row ----
    int row = tid >> 3, seg = tid & 7;         // each thread: 64 columns
    float pm = NEG_INF;
    for (int j = 0; j < 64; ++j)
        pm = fmaxf(pm, (float)Psc[(size_t)row * SDIM + seg * 64 + j]);
    red[row][seg] = pm;
    __syncthreads();
    if (seg == 0) {
        float m = red[row][0];
        for (int j = 1; j < 8; ++j) m = fmaxf(m, red[row][j]);
        rowmax[row] = m;
    }
    __syncthreads();
    float rm = rowmax[row];
    float ps = 0.0f;
    for (int j = 0; j < 64; ++j) {
        size_t idx = (size_t)row * SDIM + seg * 64 + j;
        float e = __expf((float)Psc[idx] - rm);
        Psc[idx] = (_Float16)e;
        ps += e;
    }
    red[row][seg] = ps;
    __syncthreads();
    if (seg == 0) {
        float s = 0.0f;
        for (int j = 0; j < 8; ++j) s += red[row][j];
        rowsum[row] = s;
    }
    __syncthreads();
    float inv = 1.0f / rowsum[row];
    for (int j = 0; j < 64; ++j) {
        size_t idx = (size_t)row * SDIM + seg * 64 + j;
        Psc[idx] = (_Float16)((float)Psc[idx] * inv);
    }
    __syncthreads();

    // ---- Phase 2: out 32 x 96 per wave, K = S = 512 ----
    int h0 = wave * 96;
    CV d0[6], d1[6];
    #pragma unroll
    for (int t = 0; t < 6; ++t)
        #pragma unroll
        for (int v = 0; v < 8; ++v) { d0[t].f[v] = 0.0f; d1[t].f[v] = 0.0f; }

    const _Float16* Xb = X16T + (size_t)b * HDIM * SDIM;   // [H][S] row-major
    for (int k0 = 0; k0 < SDIM; k0 += 32) {
        v16h a0 = load_matA(Psc, SDIM, k0, lane);
        v16h a1 = load_matA(Psc + (size_t)16 * SDIM, SDIM, k0, lane);
        #pragma unroll
        for (int t = 0; t < 6; ++t) {
            v16h bm = load_matB(Xb + (size_t)(h0 + t * 16) * SDIM, SDIM, k0, lane);
            d0[t].v = wmma_f16(a0, bm, d0[t].v);
            d1[t].v = wmma_f16(a1, bm, d1[t].v);
        }
    }

    float* ob = out + ((size_t)b * LDIM + l0) * HDIM;
    #pragma unroll
    for (int t = 0; t < 6; ++t)
        #pragma unroll
        for (int v = 0; v < 8; ++v) {
            int m = v + hf * 8;
            int n = h0 + t * 16 + col;
            ob[(size_t)m * HDIM + n]        = d0[t].f[v];
            ob[(size_t)(m + 16) * HDIM + n] = d1[t].f[v];
        }
}

// ---------------------------------------------------------------------------
// Launcher
// inputs: d_in[0]=inputs f32 [B,S,H], d_in[1]=masks bool [B,1,S],
//         d_in[2]=label_embedding f32 [L,H], d_in[3]=W_key f32 [H,H]
// ---------------------------------------------------------------------------
extern "C" void kernel_launch(void* const* d_in, const int* in_sizes, int n_in,
                              void* d_out, int out_size, void* d_ws, size_t ws_size,
                              hipStream_t stream) {
    const float* inputs = (const float*)d_in[0];
    const unsigned char* masks = (const unsigned char*)d_in[1]; // jnp.bool_ -> 1 byte
    const float* lab = (const float*)d_in[2];
    const float* wk  = (const float*)d_in[3];
    float* out = (float*)d_out;

    char* ws = (char*)d_ws;
    size_t off = 0;
    auto carve = [&](size_t bytes) {
        void* p = ws + off;
        off += (bytes + 255) & ~(size_t)255;
        return p;
    };
    _Float16* X16  = (_Float16*)carve((size_t)BDIM * SDIM * HDIM * 2); // f16 inputs
    _Float16* X16T = (_Float16*)carve((size_t)BDIM * HDIM * SDIM * 2); // f16 inputs^T
    _Float16* W16  = (_Float16*)carve((size_t)HDIM * HDIM * 2);        // f16 W_key
    _Float16* L16  = (_Float16*)carve((size_t)LDIM * HDIM * 2);        // f16 labels
    _Float16* K16  = (_Float16*)carve((size_t)BDIM * SDIM * HDIM * 2); // f16 attn_key
    (void)ws_size; (void)in_sizes; (void)n_in; (void)out_size;

    // casts
    {
        int n4 = (BDIM * SDIM * HDIM) / 4;
        cvt_f32_f16<<<(n4 + 255) / 256, 256, 0, stream>>>(inputs, X16, n4);
    }
    {
        int n4 = (HDIM * HDIM) / 4;
        cvt_f32_f16<<<(n4 + 255) / 256, 256, 0, stream>>>(wk, W16, n4);
    }
    {
        int n4 = (LDIM * HDIM) / 4;
        cvt_f32_f16<<<(n4 + 255) / 256, 256, 0, stream>>>(lab, L16, n4);
    }
    transpose_f16<<<(BDIM * HDIM * SDIM) / 256, 256, 0, stream>>>(inputs, X16T);

    // key projection (WMMA)
    gemm_key<<<BDIM * (SDIM / 16), 256, 0, stream>>>(X16, W16, K16);

    // fused scores + masked softmax + weighted sum (WMMA)
    label_attn<<<BDIM * (LDIM / MT), 256, 0, stream>>>(L16, K16, X16T, masks, out);
}
